// TrajectoryDecoder_22703197127134
// MI455X (gfx1250) — matrix operand order repaired and checked by reading, not compiled, not run on hardware
//
#include <hip/hip_runtime.h>

// ---------------- model constants ----------------
#define B_     8
#define NBEV_  256
#define T_     80
#define D_     256
#define H_     8
#define FF_    512
#define HID_   512
#define L_     4
#define DH_    32
#define M0_    (NBEV_ + 2)       // 258
#define MMAX_  (M0_ + T_)        // 338
#define MR_    (B_ * MMAX_)      // 2704 rows (divisible by 16)

// ---------------- WMMA / vector types ----------------
typedef __attribute__((ext_vector_type(16))) __bf16 v16bf;
typedef __attribute__((ext_vector_type(8)))  __bf16 v8bf;
typedef __attribute__((ext_vector_type(8)))  float  v8f;

__device__ __forceinline__ __bf16 f2bf(float f) {
  union { float f; unsigned u; } v; v.f = f;
  unsigned r = (v.u + 0x7fffu + ((v.u >> 16) & 1u)) >> 16;   // RNE
  unsigned short h = (unsigned short)r;
  return __builtin_bit_cast(__bf16, h);
}

#define CAT16(lo, hi) __builtin_shufflevector(lo, hi, 0, 1, 2, 3, 4, 5, 6, 7, \
                                              8, 9, 10, 11, 12, 13, 14, 15)

// ---------------- fp32 -> bf16 bulk conversion ----------------
__global__ void k_cvt(const float* __restrict__ src, __bf16* __restrict__ dst, int n) {
  int i = blockIdx.x * blockDim.x + threadIdx.x;
  if (i < n) dst[i] = f2bf(src[i]);
}

// ---------------- WMMA GEMM: out[M,N] = act(A[M,K] * W[N,K]^T + bias) ----------------
// A and W bf16. One wave per 16x64 tile (4 accumulators): A fragment loaded once per
// 32-K chunk and reused by 4 WMMAs. K fully unrolled (32 or 64 WMMAs straight-line).
// Unconditional 16-row stores (buffers are 16-row padded; M % 16 == 0 for big GEMMs,
// workspace zero-initialized so decoder padding rows are defined). No per-lane
// branches -> EXEC all-ones through every WMMA.
template<int K, int RELU, int WF, int WB>
__global__ void k_gemm(const __bf16* __restrict__ A, const __bf16* __restrict__ W,
                       const float* __restrict__ bias, float* __restrict__ C,
                       __bf16* __restrict__ Cb, int M, int N) {
  int gtid = blockIdx.x * blockDim.x + threadIdx.x;
  int wave = gtid >> 5;
  int lane = threadIdx.x & 31;
  int tilesN = N >> 6;                          // 64-wide N tiles
  int tilesM = M >> 4;
  if (wave >= tilesM * tilesN) return;          // wave-uniform exit
  int tn = wave % tilesN, tm = wave / tilesN;
  int mBase = tm << 4, nBase = tn << 6;
  int lm = lane & 15;
  int hi = lane >> 4;                           // half-wave select
  const v8bf* ap = (const v8bf*)(A + (size_t)(mBase + lm) * K + (hi << 4));
  const v8bf* bp = (const v8bf*)(W + (size_t)(nBase + lm) * K + (hi << 4));
  const size_t bstep = (size_t)16 * K / 8;      // 16 weight rows, in v8bf units
  v8f acc[4] = {{}, {}, {}, {}};
#pragma unroll
  for (int kk = 0; kk < K / 32; ++kk) {
    v16bf a  = CAT16(ap[0], ap[1]);
    v16bf b0 = CAT16(bp[0],             bp[1]);
    v16bf b1 = CAT16(bp[bstep],         bp[bstep + 1]);
    v16bf b2 = CAT16(bp[2 * bstep],     bp[2 * bstep + 1]);
    v16bf b3 = CAT16(bp[3 * bstep],     bp[3 * bstep + 1]);
    acc[0] = __builtin_amdgcn_wmma_f32_16x16x32_bf16(false, a, false, b0, (short)0, acc[0], false, false);
    acc[1] = __builtin_amdgcn_wmma_f32_16x16x32_bf16(false, a, false, b1, (short)0, acc[1], false, false);
    acc[2] = __builtin_amdgcn_wmma_f32_16x16x32_bf16(false, a, false, b2, (short)0, acc[2], false, false);
    acc[3] = __builtin_amdgcn_wmma_f32_16x16x32_bf16(false, a, false, b3, (short)0, acc[3], false, false);
    ap += 4;                                    // advance 32 elements
    bp += 4;
  }
  int rowAdd = hi << 3;                         // lanes 16-31 hold rows M+8
#pragma unroll
  for (int j = 0; j < 4; ++j) {
    float bv = bias[nBase + (j << 4) + lm];
#pragma unroll
    for (int r = 0; r < 8; ++r) {
      float v = acc[j][r] + bv;
      if (RELU) v = fmaxf(v, 0.f);
      size_t idx = (size_t)(mBase + r + rowAdd) * N + nBase + (j << 4) + lm;
      if (WF) C[idx]  = v;
      if (WB) Cb[idx] = f2bf(v);
    }
  }
}

// ---------------- encoder self-attention: one wave per (b,h,q); lane == head-dim ----------------
// qkv fp32 [B*S,768]; emits bf16 (feeds the output-projection GEMM only).
__global__ void k_attn(const float* __restrict__ qkv, __bf16* __restrict__ out,
                       int S, int valid, float scale) {
  int gtid = blockIdx.x * blockDim.x + threadIdx.x;
  int wave = gtid >> 5, lane = threadIdx.x & 31;
  if (wave >= B_ * H_ * S) return;
  int qpos = wave % S; int bh = wave / S; int h = bh % H_; int b = bh / H_;
  const float* base = qkv + (size_t)b * S * 768;
  float qv = base[(size_t)qpos * 768 + h * DH_ + lane];
  float mx = -3.0e38f, l = 0.f, acc = 0.f;
  for (int k = 0; k < valid; ++k) {
    const float* kr = base + (size_t)k * 768 + 256 + h * DH_;
    float p = qv * kr[lane];
#pragma unroll
    for (int msk = 16; msk > 0; msk >>= 1) p += __shfl_xor(p, msk, 32);
    float s  = p * scale;
    float nm = fmaxf(mx, s);
    float c  = __expf(mx - nm);
    float e  = __expf(s - nm);
    float v  = base[(size_t)k * 768 + 512 + h * DH_ + lane];
    l   = l * c + e;
    acc = acc * c + e * v;
    mx  = nm;
  }
  out[((size_t)(b * S + qpos)) * D_ + h * DH_ + lane] = f2bf(acc / l);
}

// ---------------- decoder cross-attention: one wave per (b,h) ----------------
__global__ void k_xattn(const float* __restrict__ qb, const float* __restrict__ kv,
                        __bf16* __restrict__ out, int valid, float scale) {
  int gtid = blockIdx.x * blockDim.x + threadIdx.x;
  int wave = gtid >> 5, lane = threadIdx.x & 31;
  if (wave >= B_ * H_) return;
  int h = wave % H_, b = wave / H_;
  float qv = qb[b * D_ + h * DH_ + lane];
  const float* base = kv + (size_t)b * MMAX_ * 512;
  float mx = -3.0e38f, l = 0.f, acc = 0.f;
  for (int k = 0; k < valid; ++k) {
    const float* kr = base + (size_t)k * 512 + h * DH_;
    float p = qv * kr[lane];
#pragma unroll
    for (int msk = 16; msk > 0; msk >>= 1) p += __shfl_xor(p, msk, 32);
    float s  = p * scale;
    float nm = fmaxf(mx, s);
    float c  = __expf(mx - nm);
    float e  = __expf(s - nm);
    float v  = kr[256 + lane];
    l   = l * c + e;
    acc = acc * c + e * v;
    mx  = nm;
  }
  out[b * D_ + h * DH_ + lane] = f2bf(acc / l);
}

// ---------------- LayerNorm (+optional residual), dual fp32/bf16 output ----------------
__global__ void k_ln(const float* __restrict__ x, const float* __restrict__ res,
                     const float* __restrict__ g, const float* __restrict__ bta,
                     float* __restrict__ outF, __bf16* __restrict__ outB, int R) {
  int gtid = blockIdx.x * blockDim.x + threadIdx.x;
  int wave = gtid >> 5, lane = threadIdx.x & 31;
  if (wave >= R) return;
  const float* xr = x + (size_t)wave * D_;
  const float* rr = res ? res + (size_t)wave * D_ : nullptr;
  float v[8], s = 0.f;
#pragma unroll
  for (int i = 0; i < 8; ++i) {
    int d = i * 32 + lane;
    float t = xr[d] + (rr ? rr[d] : 0.f);
    v[i] = t; s += t;
  }
#pragma unroll
  for (int m = 16; m > 0; m >>= 1) s += __shfl_xor(s, m, 32);
  float mean = s * (1.0f / 256.0f), q = 0.f;
#pragma unroll
  for (int i = 0; i < 8; ++i) { float c = v[i] - mean; q += c * c; }
#pragma unroll
  for (int m = 16; m > 0; m >>= 1) q += __shfl_xor(q, m, 32);
  float inv = rsqrtf(q * (1.0f / 256.0f) + 1e-5f);
#pragma unroll
  for (int i = 0; i < 8; ++i) {
    int d = i * 32 + lane;
    float o = (v[i] - mean) * inv * g[d] + bta[d];
    outF[(size_t)wave * D_ + d] = o;
    if (outB) outB[(size_t)wave * D_ + d] = f2bf(o);
  }
}

// ---------------- embed_frame (dual output) ----------------
__global__ void k_embed(const float* __restrict__ src3, const float* __restrict__ fW,
                        const float* __restrict__ fb, const float* __restrict__ te,
                        const float* __restrict__ pe, int t,
                        float* __restrict__ dst, int dstStride,
                        __bf16* __restrict__ dstB) {
  int i = blockIdx.x * blockDim.x + threadIdx.x;
  if (i >= B_ * D_) return;
  int d = i & (D_ - 1), b = i >> 8;
  const float* s = src3 + b * 3;
  float v = fW[d * 3 + 0] * s[0] + fW[d * 3 + 1] * s[1] + fW[d * 3 + 2] * s[2]
          + fb[d] + te[(size_t)t * D_ + d] + pe[(size_t)t * D_ + d];
  dst[(size_t)b * dstStride + d] = v;
  if (dstB) dstB[(size_t)b * D_ + d] = f2bf(v);
}

// ---------------- decoder self-attn over 1 token == copy V (emit bf16) ----------------
__global__ void k_copy_v(const float* __restrict__ dqkv, __bf16* __restrict__ datt) {
  int i = blockIdx.x * blockDim.x + threadIdx.x;
  if (i >= B_ * D_) return;
  int d = i & (D_ - 1), b = i >> 8;
  datt[b * D_ + d] = f2bf(dqkv[b * 768 + 512 + d]);
}

// ---------------- head layer 2: pred[b,o] = hid[b]@W2[o] + b2[o] ----------------
__global__ void k_head2(const float* __restrict__ hid, const float* __restrict__ W2,
                        const float* __restrict__ b2, float* __restrict__ pred,
                        float* __restrict__ outp) {
  int gtid = blockIdx.x * blockDim.x + threadIdx.x;
  int wave = gtid >> 5, lane = threadIdx.x & 31;
  if (wave >= B_ * 3) return;
  int o = wave % 3, b = wave / 3;
  const float* hr = hid + (size_t)b * HID_;
  const float* wr = W2 + (size_t)o * HID_;
  float p = 0.f;
  for (int j = lane; j < HID_; j += 32) p += hr[j] * wr[j];
#pragma unroll
  for (int m = 16; m > 0; m >>= 1) p += __shfl_xor(p, m, 32);
  if (lane == 0) {
    float v = p + b2[o];
    pred[b * 3 + o] = v;
    outp[(size_t)b * T_ * 3 + o] = v;
  }
}

// ---------------- memory init: zeros + bev/text/pose ----------------
__global__ void k_init_mem(const float* __restrict__ bev, const float* __restrict__ text,
                           const float* __restrict__ pose, float* __restrict__ mem) {
  int i = blockIdx.x * blockDim.x + threadIdx.x;
  if (i >= B_ * MMAX_ * D_) return;
  int d = i & (D_ - 1);
  int bs = i >> 8;
  int s = bs % MMAX_, b = bs / MMAX_;
  float v = 0.f;
  if (s < NBEV_)           v = bev[((size_t)b * NBEV_ + s) * D_ + d];
  else if (s == NBEV_)     v = text[b * D_ + d];
  else if (s == NBEV_ + 1) v = pose[b * D_ + d];
  mem[(size_t)i] = v;
}

// ================= host orchestration =================
enum {
  IN_BEV = 0, IN_TEXT, IN_POSE, IN_START, IN_TF,
  IN_TIME_EMB, IN_FRAME_W, IN_FRAME_B, IN_HW1, IN_HB1, IN_HW2, IN_HB2,
  IN_E_WQKV, IN_E_BQKV, IN_E_WO, IN_E_BO, IN_E_W1, IN_E_B1, IN_E_W2, IN_E_B2,
  IN_E_LN1G, IN_E_LN1B, IN_E_LN2G, IN_E_LN2B,
  IN_D_WQKV, IN_D_BQKV, IN_D_WO, IN_D_BO,
  IN_D_CWQKV, IN_D_CBQKV, IN_D_CWO, IN_D_CBO,
  IN_D_W1, IN_D_B1, IN_D_W2, IN_D_B2,
  IN_D_LN1G, IN_D_LN1B, IN_D_LN2G, IN_D_LN2B, IN_D_LN3G, IN_D_LN3B,
  IN_ENC_NG, IN_ENC_NB, IN_DEC_NG, IN_DEC_NB, IN_POS_EMB
};

static inline void launch_gemm(const __bf16* A, const __bf16* W, const float* bias,
                               float* C, __bf16* Cb, int M, int N, int K, int relu,
                               hipStream_t s) {
  int tiles = (M >> 4) * (N >> 6);
  dim3 g((tiles * 32 + 255) / 256), b(256);
  if (K == 256) {
    if (relu) {
      if (Cb) k_gemm<256, 1, 0, 1><<<g, b, 0, s>>>(A, W, bias, C, Cb, M, N);
      else    k_gemm<256, 1, 1, 0><<<g, b, 0, s>>>(A, W, bias, C, Cb, M, N);
    } else    k_gemm<256, 0, 1, 0><<<g, b, 0, s>>>(A, W, bias, C, Cb, M, N);
  } else      k_gemm<512, 0, 1, 0><<<g, b, 0, s>>>(A, W, bias, C, Cb, M, N);
}

extern "C" void kernel_launch(void* const* d_in, const int* in_sizes, int n_in,
                              void* d_out, int out_size, void* d_ws, size_t ws_size,
                              hipStream_t stream) {
  (void)in_sizes; (void)out_size; (void)ws_size;
  const float* F[47];
  for (int i = 0; i < 47 && i < n_in; ++i) F[i] = (const float*)d_in[i];

  // ---- workspace layout ----
  size_t off = 0;
  auto take = [&](size_t bytes) { size_t o = off; off = (off + bytes + 255) & ~(size_t)255; return o; };
  auto fptr = [&](size_t o) { return (float*)((char*)d_ws + o); };
  auto bptr = [&](size_t o) { return (__bf16*)((char*)d_ws + o); };

  const size_t SZ_WQKV = (size_t)L_ * 768 * 256, SZ_WO = (size_t)L_ * 256 * 256,
               SZ_W1 = (size_t)L_ * 512 * 256,   SZ_W2 = (size_t)L_ * 256 * 512;
  size_t o_eWqkv = take(SZ_WQKV * 2), o_eWo = take(SZ_WO * 2),
         o_eW1 = take(SZ_W1 * 2),     o_eW2 = take(SZ_W2 * 2);
  size_t o_dWqkv = take(SZ_WQKV * 2), o_dWo = take(SZ_WO * 2),
         o_cWqkv = take(SZ_WQKV * 2), o_cWo = take(SZ_WO * 2),
         o_dW1 = take(SZ_W1 * 2),     o_dW2 = take(SZ_W2 * 2),
         o_hW1 = take((size_t)HID_ * 256 * 2);
  size_t o_mem   = take((size_t)MR_ * D_ * 4);
  size_t o_x     = take((size_t)MR_ * D_ * 4);
  size_t o_xb    = take((size_t)MR_ * D_ * 2);
  size_t o_qkv   = take((size_t)MR_ * 768 * 4);
  size_t o_attb  = take((size_t)MR_ * D_ * 2);
  size_t o_tmp   = take((size_t)MR_ * D_ * 4);
  size_t o_ffnb  = take((size_t)MR_ * FF_ * 2);
  size_t o_enc   = take((size_t)MR_ * D_ * 4);
  size_t o_encb  = take((size_t)MR_ * D_ * 2);
  size_t o_kv    = take((size_t)MR_ * 512 * 4);
  size_t o_dq    = take(16 * D_ * 4);
  size_t o_dqb   = take(16 * D_ * 2);
  size_t o_dqkv  = take(16 * 768 * 4);
  size_t o_dattb = take(16 * D_ * 2);
  size_t o_dtmp  = take(16 * D_ * 4);
  size_t o_dhid  = take(16 * HID_ * 4);
  size_t o_dhidb = take(16 * HID_ * 2);
  size_t o_dnorm = take(16 * D_ * 4);
  size_t o_dnormb= take(16 * D_ * 2);
  size_t o_qb    = take(16 * D_ * 4);
  size_t o_pred  = take(32 * 4);
  size_t total   = off;

  float  *mem = fptr(o_mem), *x = fptr(o_x), *qkv = fptr(o_qkv), *tmp = fptr(o_tmp),
         *enc = fptr(o_enc), *kv = fptr(o_kv), *dq = fptr(o_dq), *dqkv = fptr(o_dqkv),
         *dtmp = fptr(o_dtmp), *dhid = fptr(o_dhid), *dnorm = fptr(o_dnorm),
         *qb = fptr(o_qb), *pred = fptr(o_pred);
  __bf16 *xb = bptr(o_xb), *attb = bptr(o_attb), *ffnb = bptr(o_ffnb), *encb = bptr(o_encb),
         *dqb = bptr(o_dqb), *dattb = bptr(o_dattb), *dhidb = bptr(o_dhidb),
         *dnormb = bptr(o_dnormb);
  __bf16 *eWqkv = bptr(o_eWqkv), *eWo = bptr(o_eWo), *eW1 = bptr(o_eW1), *eW2 = bptr(o_eW2),
         *dWqkv = bptr(o_dWqkv), *dWo = bptr(o_dWo), *cWqkv = bptr(o_cWqkv), *cWo = bptr(o_cWo),
         *dW1 = bptr(o_dW1), *dW2 = bptr(o_dW2), *hW1 = bptr(o_hW1);

  auto CVT = [&](const float* s, __bf16* d, size_t n) {
    k_cvt<<<dim3((unsigned)((n + 255) / 256)), dim3(256), 0, stream>>>(s, d, (int)n);
  };
  auto LN = [&](const float* xx, const float* rr, const float* g, const float* b,
                float* oF, __bf16* oB, int R) {
    k_ln<<<dim3((R * 32 + 255) / 256), dim3(256), 0, stream>>>(xx, rr, g, b, oF, oB, R);
  };

  // ---- per-launch init: zero scratch (defines decoder padding rows), convert weights ----
  hipMemsetAsync(d_ws, 0, total, stream);
  CVT(F[IN_E_WQKV], eWqkv, SZ_WQKV);  CVT(F[IN_E_WO], eWo, SZ_WO);
  CVT(F[IN_E_W1],  eW1,  SZ_W1);      CVT(F[IN_E_W2], eW2, SZ_W2);
  CVT(F[IN_D_WQKV], dWqkv, SZ_WQKV);  CVT(F[IN_D_WO], dWo, SZ_WO);
  CVT(F[IN_D_CWQKV], cWqkv, SZ_WQKV); CVT(F[IN_D_CWO], cWo, SZ_WO);
  CVT(F[IN_D_W1], dW1, SZ_W1);        CVT(F[IN_D_W2], dW2, SZ_W2);
  CVT(F[IN_HW1], hW1, (size_t)HID_ * 256);
  k_init_mem<<<dim3((B_ * MMAX_ * D_ + 255) / 256), dim3(256), 0, stream>>>(
      F[IN_BEV], F[IN_TEXT], F[IN_POSE], mem);

  const float scale = 0.17677669529663687f;  // 1/sqrt(DH)
  float* outF = (float*)d_out;

  for (int t = 1; t <= T_; ++t) {
    int valid = M0_ + t - 1;
    // ----- encoder over memory -----
    hipMemcpyAsync(x, mem, (size_t)MR_ * D_ * 4, hipMemcpyDeviceToDevice, stream);
    CVT(mem, xb, (size_t)MR_ * D_);
    for (int i = 0; i < L_; ++i) {
      launch_gemm(xb, eWqkv + (size_t)i * 768 * 256, F[IN_E_BQKV] + i * 768,
                  qkv, nullptr, MR_, 768, 256, 0, stream);
      k_attn<<<dim3((B_ * H_ * MMAX_ * 32 + 255) / 256), dim3(256), 0, stream>>>(
          qkv, attb, MMAX_, valid, scale);
      launch_gemm(attb, eWo + (size_t)i * 256 * 256, F[IN_E_BO] + i * 256,
                  tmp, nullptr, MR_, 256, 256, 0, stream);
      LN(x, tmp, F[IN_E_LN1G] + i * 256, F[IN_E_LN1B] + i * 256, x, xb, MR_);
      launch_gemm(xb, eW1 + (size_t)i * 512 * 256, F[IN_E_B1] + i * 512,
                  nullptr, ffnb, MR_, 512, 256, 1, stream);
      launch_gemm(ffnb, eW2 + (size_t)i * 256 * 512, F[IN_E_B2] + i * 256,
                  tmp, nullptr, MR_, 256, 512, 0, stream);
      LN(x, tmp, F[IN_E_LN2G] + i * 256, F[IN_E_LN2B] + i * 256, x, xb, MR_);
    }
    LN(x, nullptr, F[IN_ENC_NG], F[IN_ENC_NB], enc, encb, MR_);

    // ----- decoder (single query token, buffers padded to 16 rows) -----
    const float* prevPtr = (t == 1) ? F[IN_START] : pred;
    k_embed<<<dim3((B_ * D_ + 255) / 256), dim3(256), 0, stream>>>(
        prevPtr, F[IN_FRAME_W], F[IN_FRAME_B], F[IN_TIME_EMB], F[IN_POS_EMB], t,
        dq, D_, dqb);
    for (int i = 0; i < L_; ++i) {
      // self-attn over 1 token: attn(x) == V
      launch_gemm(dqb, dWqkv + (size_t)i * 768 * 256, F[IN_D_BQKV] + i * 768,
                  dqkv, nullptr, 16, 768, 256, 0, stream);
      k_copy_v<<<dim3((B_ * D_ + 255) / 256), dim3(256), 0, stream>>>(dqkv, dattb);
      launch_gemm(dattb, dWo + (size_t)i * 256 * 256, F[IN_D_BO] + i * 256,
                  dtmp, nullptr, 16, 256, 256, 0, stream);
      LN(dq, dtmp, F[IN_D_LN1G] + i * 256, F[IN_D_LN1B] + i * 256, dq, dqb, B_);
      // cross-attn
      launch_gemm(dqb, cWqkv + (size_t)i * 768 * 256, F[IN_D_CBQKV] + i * 768,
                  qb, nullptr, 16, 256, 256, 0, stream);
      launch_gemm(encb, cWqkv + (size_t)i * 768 * 256 + 256 * 256,
                  F[IN_D_CBQKV] + i * 768 + 256, kv, nullptr, MR_, 512, 256, 0, stream);
      k_xattn<<<dim3((B_ * H_ * 32 + 255) / 256), dim3(256), 0, stream>>>(
          qb, kv, dattb, valid, scale);
      launch_gemm(dattb, cWo + (size_t)i * 256 * 256, F[IN_D_CBO] + i * 256,
                  dtmp, nullptr, 16, 256, 256, 0, stream);
      LN(dq, dtmp, F[IN_D_LN2G] + i * 256, F[IN_D_LN2B] + i * 256, dq, dqb, B_);
      // FFN
      launch_gemm(dqb, dW1 + (size_t)i * 512 * 256, F[IN_D_B1] + i * 512,
                  nullptr, dhidb, 16, 512, 256, 1, stream);
      launch_gemm(dhidb, dW2 + (size_t)i * 256 * 512, F[IN_D_B2] + i * 256,
                  dtmp, nullptr, 16, 256, 512, 0, stream);
      LN(dq, dtmp, F[IN_D_LN3G] + i * 256, F[IN_D_LN3B] + i * 256, dq, dqb, B_);
    }
    LN(dq, nullptr, F[IN_DEC_NG], F[IN_DEC_NB], dnorm, dnormb, B_);

    // ----- head + state update -----
    launch_gemm(dnormb, hW1, F[IN_HB1], dhid, nullptr, 16, HID_, 256, 1, stream);
    k_head2<<<dim3(3), dim3(256), 0, stream>>>(dhid, F[IN_HW2], F[IN_HB2], pred,
                                               outF + (size_t)(t - 1) * 3);
    k_embed<<<dim3((B_ * D_ + 255) / 256), dim3(256), 0, stream>>>(
        pred, F[IN_FRAME_W], F[IN_FRAME_B], F[IN_TIME_EMB], F[IN_POS_EMB], t,
        mem + (size_t)(M0_ + t - 1) * D_, MMAX_ * D_, nullptr);
  }
}
